// MambaBackbone_1864015806894
// MI455X (gfx1250) — matrix operand order repaired and checked by reading, not compiled, not run on hardware
//
#include <hip/hip_runtime.h>
#include <hip/hip_bf16.h>
#include <stdint.h>

// ---------------- problem sizes ----------------
#define BATCH   32
#define LSEQ    2048
#define D_IN    8
#define D_MODEL 256
#define DEPTH   4
#define D_INNER 512
#define N_STATE 16
#define DT_RANK 16
#define D_CONV  4
#define BL      (BATCH * LSEQ)   // 65536 rows

// ---------------- WMMA / TDM types (gfx1250, wave32) ----------------
typedef __attribute__((ext_vector_type(16))) __bf16        v16bf;
typedef __attribute__((ext_vector_type(8)))  float         v8f;
typedef __attribute__((ext_vector_type(4)))  unsigned int  v4u;
typedef __attribute__((ext_vector_type(8)))  int           v8i;
typedef __attribute__((ext_vector_type(4)))  int           v4i;

union FragU { v16bf v; uint32_t u[8]; };

// LDS tile row stride in bf16 elements: 32 data + 8 pad = 40 (20 u32, 16B-aligned rows)
#define LSTR   40
#define LSTR32 20

// RNE float -> bf16 bits
__device__ __forceinline__ uint16_t f32_to_bf16(float f) {
    uint32_t u = __float_as_uint(f);
    if ((u & 0x7FFFFFFFu) > 0x7F800000u)
        return (uint16_t)((u | 0x00400000u) >> 16);      // NaN -> quiet
    uint32_t r = u + 0x7FFFu + ((u >> 16) & 1u);
    return (uint16_t)(r >> 16);
}

__device__ __forceinline__ float fast_silu(float v) {
    return __fdividef(v, 1.0f + __expf(-v));
}

// =====================================================================
// Tensor Data Mover: DMA a 2D bf16 tile [rows x cols] (row stride
// `stride` elements) from global into LDS at byte offset lds_off,
// padding each 16-DWORD row with 4 DWORDs => LDS row stride 40 halves.
// D# per ISA 8.3/8.4; this toolchain's 6-arg builtin; tracked by TENSORcnt.
// =====================================================================
__device__ __forceinline__ void tdm_load_2d_bf16(uint32_t lds_off,
                                                 const uint16_t* gptr,
                                                 uint32_t rows, uint32_t cols,
                                                 uint32_t stride) {
    const uint64_t ga = (uint64_t)(uintptr_t)gptr;
    v4u g0;
    g0[0] = 1u;                                          // count=1, user mode
    g0[1] = lds_off;                                     // lds_addr [63:32]
    g0[2] = (uint32_t)(ga & 0xFFFFFFFFu);                // global_addr lo
    g0[3] = (uint32_t)((ga >> 32) & 0x01FFFFFFu)         // global_addr [56:32]
          | (2u << 30);                                  // type=2 (image)
    v8i g1;
    // data_size=1(2B) | pad_enable | pad_interval=3 (16 DW) | pad_amount=3 (4 DW)
    g1[0] = (int)((1u << 16) | (1u << 20) | (3u << 22) | (3u << 25));
    g1[1] = (int)((cols & 0xFFFFu) << 16);               // tensor_dim0 lo16
    g1[2] = (int)((cols >> 16) | ((rows & 0xFFFFu) << 16));   // dim0 hi | dim1 lo
    g1[3] = (int)((rows >> 16) | ((cols & 0xFFFFu) << 16));   // dim1 hi | tile_dim0
    g1[4] = (int)(rows & 0xFFFFu);                       // tile_dim1 (tile_dim2=0)
    g1[5] = (int)stride;                                 // tensor_dim0_stride lo32
    g1[6] = 0;
    g1[7] = 0;
    const v4i gz4 = {0, 0, 0, 0};
    const v8i gz8 = {0, 0, 0, 0, 0, 0, 0, 0};
    // 6-arg form (clang-23 / therock-10.0 headers lane)
    __builtin_amdgcn_tensor_load_to_lds(g0, g1, gz4, gz4, gz8, 0);
}

// =====================================================================
// generic f32 -> bf16 converter (weights)
// =====================================================================
__global__ void k_f32_to_bf16(const float* __restrict__ src,
                              uint16_t* __restrict__ dst, int n) {
    int i = blockIdx.x * 256 + threadIdx.x;
    if (i < n) dst[i] = f32_to_bf16(src[i]);
}

// =====================================================================
// input projection: h[bl,d] = x[bl,:8] . in_w[d,:8] + in_b[d]
// =====================================================================
__global__ __launch_bounds__(256)
void k_in_proj(const float* __restrict__ x, const float* __restrict__ in_w,
               const float* __restrict__ in_b,
               float* __restrict__ h, uint16_t* __restrict__ hb) {
    const int row = blockIdx.x;
    const int d   = threadIdx.x;
    const float* xr = x + (size_t)row * D_IN;
    const float* w  = in_w + (size_t)d * D_IN;
    float acc = in_b[d];
#pragma unroll
    for (int i = 0; i < D_IN; ++i) acc += w[i] * xr[i];
    h [(size_t)row * D_MODEL + d] = acc;
    hb[(size_t)row * D_MODEL + d] = f32_to_bf16(acc);
}

// =====================================================================
// WMMA bf16 GEMM:  C[M x N] = A[M x K](bf16) * W[N x K](bf16)^T
// block = 256 threads (8 waves), tile = 128 (M) x NT*16 (N).
// K-stages double-buffered in LDS via tensor_load_to_lds (wave 0 issues,
// s_wait_tensorcnt<=2 exploits in-order TDM completion).
// Frag layouts per ISA 7.12.2 (bf16 A 16x32 / B 32x16, f32 C/D 16x16).
// =====================================================================
template <int NT>
__global__ __launch_bounds__(256)
void k_gemm_bf16(const uint16_t* __restrict__ A,   // [M x K] bf16 bits
                 const uint16_t* __restrict__ W,   // [N x K] bf16 bits
                 float* __restrict__ C,            // [M x N] f32 (or null)
                 uint16_t* __restrict__ Cb,        // [M x N] bf16 (or null)
                 int K, int N) {
    constexpr int BN = NT * 16;
    __shared__ __align__(16) uint16_t As[2][128 * LSTR];
    __shared__ __align__(16) uint16_t Ws[2][BN  * LSTR];

    const int tid  = threadIdx.x;
    const int wid  = tid >> 5;
    const int lane = tid & 31;
    const int ln16 = lane & 15;
    const int khal = lane >> 4;
    const int m0 = blockIdx.x * 128;
    const int n0 = blockIdx.y * BN;

    v8f acc[NT];
#pragma unroll
    for (int t = 0; t < NT; ++t)
#pragma unroll
        for (int i = 0; i < 8; ++i) acc[t][i] = 0.0f;

    const int  nk     = K >> 5;
    const bool issuer = (tid < 32);          // wave 0 drives the TDM

    if (issuer) {
        tdm_load_2d_bf16((uint32_t)(uintptr_t)&As[0][0], A + (size_t)m0 * K,
                         128, 32, (uint32_t)K);
        tdm_load_2d_bf16((uint32_t)(uintptr_t)&Ws[0][0], W + (size_t)n0 * K,
                         BN, 32, (uint32_t)K);
    }

    for (int i = 0; i < nk; ++i) {
        const int cur = i & 1;
        if ((i + 1 < nk) && issuer) {        // prefetch next K-stage
            const int k0n = (i + 1) << 5;
            tdm_load_2d_bf16((uint32_t)(uintptr_t)&As[cur ^ 1][0],
                             A + (size_t)m0 * K + k0n, 128, 32, (uint32_t)K);
            tdm_load_2d_bf16((uint32_t)(uintptr_t)&Ws[cur ^ 1][0],
                             W + (size_t)n0 * K + k0n, BN, 32, (uint32_t)K);
        }
        if (issuer) {
            if (i + 1 < nk) __builtin_amdgcn_s_wait_tensorcnt(2);
            else            __builtin_amdgcn_s_wait_tensorcnt(0);
        }
        __syncthreads();

        const uint32_t* As32 = (const uint32_t*)&As[cur][0];
        const uint32_t* Ws32 = (const uint32_t*)&Ws[cur][0];

        // A fragment: 2x aligned b128 LDS loads
        FragU af;
        {
            const int rb = (wid * 16 + ln16) * LSTR32 + (khal << 2);
            const uint4 a0 = *(const uint4*)&As32[rb];
            const uint4 a1 = *(const uint4*)&As32[rb + 8];
            af.u[0] = a0.x; af.u[1] = a0.y; af.u[2] = a0.z; af.u[3] = a0.w;
            af.u[4] = a1.x; af.u[5] = a1.y; af.u[6] = a1.z; af.u[7] = a1.w;
        }
#pragma unroll
        for (int t = 0; t < NT; ++t) {
            FragU bw;
            const int rb = (t * 16 + ln16) * LSTR32 + (khal << 2);
            const uint4 b0 = *(const uint4*)&Ws32[rb];
            const uint4 b1 = *(const uint4*)&Ws32[rb + 8];
            bw.u[0] = b0.x; bw.u[1] = b0.y; bw.u[2] = b0.z; bw.u[3] = b0.w;
            bw.u[4] = b1.x; bw.u[5] = b1.y; bw.u[6] = b1.z; bw.u[7] = b1.w;
            acc[t] = __builtin_amdgcn_wmma_f32_16x16x32_bf16(
                false, af.v, false, bw.v, (short)0, acc[t], false, false);
        }
        __syncthreads();
    }

    // epilogue: C/D layout (VGPR r -> M = r + 8*khal, N = ln16)
#pragma unroll
    for (int t = 0; t < NT; ++t) {
#pragma unroll
        for (int r = 0; r < 8; ++r) {
            int row = m0 + wid * 16 + r + 8 * khal;
            int col = n0 + t * 16 + ln16;
            float v = acc[t][r];
            if (C)  C [(size_t)row * N + col] = v;
            if (Cb) Cb[(size_t)row * N + col] = f32_to_bf16(v);
        }
    }
}

// =====================================================================
// dt projection (K=16 zero-padded to 32) + bias + fast softplus, WMMA.
// A = dbc[:, 0:16] converted f32->bf16 while staging.
// =====================================================================
__global__ __launch_bounds__(256)
void k_dt(const float* __restrict__ dbc,        // [BL x 48]
          const uint16_t* __restrict__ Wdt,     // [512 x 16] bf16
          const float* __restrict__ dt_bias,    // [512]
          float* __restrict__ dt_out) {         // [BL x 512]
    __shared__ __align__(16) uint16_t As[128 * LSTR];
    __shared__ __align__(16) uint16_t Ws[64 * LSTR];
    const int tid  = threadIdx.x;
    const int wid  = tid >> 5;
    const int lane = tid & 31;
    const int ln16 = lane & 15;
    const int khal = lane >> 4;
    const int m0 = blockIdx.x * 128;
    const int n0 = blockIdx.y * 64;

    for (int i = tid; i < 128 * 32; i += 256) {
        int r = i >> 5, k = i & 31;
        As[r * LSTR + k] = (k < DT_RANK)
            ? f32_to_bf16(dbc[(size_t)(m0 + r) * 48 + k]) : (uint16_t)0;
    }
    for (int i = tid; i < 64 * 32; i += 256) {
        int r = i >> 5, k = i & 31;
        Ws[r * LSTR + k] = (k < DT_RANK) ? Wdt[(n0 + r) * DT_RANK + k]
                                         : (uint16_t)0;
    }
    __syncthreads();

    const uint32_t* As32 = (const uint32_t*)As;
    const uint32_t* Ws32 = (const uint32_t*)Ws;

    FragU af;
    {
        const int rb = (wid * 16 + ln16) * LSTR32 + (khal << 2);
        const uint4 a0 = *(const uint4*)&As32[rb];
        const uint4 a1 = *(const uint4*)&As32[rb + 8];
        af.u[0] = a0.x; af.u[1] = a0.y; af.u[2] = a0.z; af.u[3] = a0.w;
        af.u[4] = a1.x; af.u[5] = a1.y; af.u[6] = a1.z; af.u[7] = a1.w;
    }
    v8f acc[4];
#pragma unroll
    for (int t = 0; t < 4; ++t) {
#pragma unroll
        for (int i = 0; i < 8; ++i) acc[t][i] = 0.0f;
        FragU bw;
        const int rb = (t * 16 + ln16) * LSTR32 + (khal << 2);
        const uint4 b0 = *(const uint4*)&Ws32[rb];
        const uint4 b1 = *(const uint4*)&Ws32[rb + 8];
        bw.u[0] = b0.x; bw.u[1] = b0.y; bw.u[2] = b0.z; bw.u[3] = b0.w;
        bw.u[4] = b1.x; bw.u[5] = b1.y; bw.u[6] = b1.z; bw.u[7] = b1.w;
        acc[t] = __builtin_amdgcn_wmma_f32_16x16x32_bf16(
            false, af.v, false, bw.v, (short)0, acc[t], false, false);
    }
#pragma unroll
    for (int t = 0; t < 4; ++t) {
#pragma unroll
        for (int r = 0; r < 8; ++r) {
            int row = m0 + wid * 16 + r + 8 * khal;
            int col = n0 + t * 16 + ln16;
            float v  = acc[t][r] + dt_bias[col];
            float sp = (v > 20.0f) ? v : __logf(1.0f + __expf(v)); // softplus
            dt_out[(size_t)row * D_INNER + col] = sp;
        }
    }
}

// =====================================================================
// depthwise causal conv (width 4) + SiLU over xpart = xz[:, :512]
// =====================================================================
__global__ __launch_bounds__(512)
void k_conv(const float* __restrict__ xz, const float* __restrict__ cw,
            const float* __restrict__ cb,
            float* __restrict__ xc, uint16_t* __restrict__ xcb) {
    const int bl = blockIdx.x;
    const int l  = bl & (LSEQ - 1);
    const int d  = threadIdx.x;
    const float* w = cw + (size_t)d * D_CONV;
    float acc = cb[d];
#pragma unroll
    for (int j = 0; j < D_CONV; ++j) {
        int lt = l + j - (D_CONV - 1);
        if (lt >= 0)
            acc += w[j] * xz[(size_t)(bl + j - (D_CONV - 1)) * (2 * D_INNER) + d];
    }
    float s = fast_silu(acc);
    xc [(size_t)bl * D_INNER + d] = s;
    xcb[(size_t)bl * D_INNER + d] = f32_to_bf16(s);
}

// =====================================================================
// selective scan fused with +u*D and SiLU(z) gate; one thread per (b,d),
// 16-wide f32 state in registers, recurrence exp via v_exp_f32.
// =====================================================================
__global__ __launch_bounds__(128)
void k_scan(const float* __restrict__ xc,  const float* __restrict__ dtv,
            const float* __restrict__ dbc, const float* __restrict__ xz,
            const float* __restrict__ A_log, const float* __restrict__ Dp,
            uint16_t* __restrict__ y2b) {
    const int b = blockIdx.x;
    const int d = blockIdx.y * 128 + threadIdx.x;
    float Ac[N_STATE], hst[N_STATE];
#pragma unroll
    for (int n = 0; n < N_STATE; ++n) {
        Ac[n]  = -__expf(A_log[(size_t)d * N_STATE + n]);
        hst[n] = 0.0f;
    }
    const float Dv = Dp[d];
    const size_t base = (size_t)b * LSEQ;
    for (int t = 0; t < LSEQ; ++t) {
        const size_t row = base + t;
        const float u  = xc [row * D_INNER + d];
        const float dt = dtv[row * D_INNER + d];
        const float* bc = dbc + row * 48;            // broadcast across block
        const float du = dt * u;
        float y = 0.0f;
#pragma unroll
        for (int n = 0; n < N_STATE; ++n) {
            float dA = __expf(dt * Ac[n]);
            hst[n] = dA * hst[n] + du * bc[DT_RANK + n];
            y += hst[n] * bc[DT_RANK + N_STATE + n];
        }
        const float z = xz[row * (2 * D_INNER) + D_INNER + d];
        y2b[row * D_INNER + d] = f32_to_bf16((y + u * Dv) * fast_silu(z));
    }
}

// =====================================================================
// final LayerNorm over D_MODEL=256
// =====================================================================
__global__ __launch_bounds__(256)
void k_ln(const float* __restrict__ h, const float* __restrict__ g,
          const float* __restrict__ b, float* __restrict__ out) {
    __shared__ float red[2][8];
    const int row = blockIdx.x, tid = threadIdx.x;
    const float v = h[(size_t)row * D_MODEL + tid];
    float s = v, s2 = v * v;
#pragma unroll
    for (int off = 16; off > 0; off >>= 1) {
        s  += __shfl_down(s,  off, 32);
        s2 += __shfl_down(s2, off, 32);
    }
    const int wid = tid >> 5, lane = tid & 31;
    if (lane == 0) { red[0][wid] = s; red[1][wid] = s2; }
    __syncthreads();
    if (tid == 0) {
        float ts = 0.f, ts2 = 0.f;
        for (int i = 0; i < 8; ++i) { ts += red[0][i]; ts2 += red[1][i]; }
        red[0][0] = ts; red[1][0] = ts2;
    }
    __syncthreads();
    const float mean = red[0][0] * (1.0f / D_MODEL);
    const float var  = red[1][0] * (1.0f / D_MODEL) - mean * mean;
    out[(size_t)row * D_MODEL + tid] =
        (v - mean) * rsqrtf(var + 1e-5f) * g[tid] + b[tid];
}

// =====================================================================
// host launcher
// =====================================================================
extern "C" void kernel_launch(void* const* d_in, const int* in_sizes, int n_in,
                              void* d_out, int out_size, void* d_ws, size_t ws_size,
                              hipStream_t stream) {
    const float* x      = (const float*)d_in[0];
    const float* in_w   = (const float*)d_in[1];
    const float* in_b   = (const float*)d_in[2];
    const float* m_in_w = (const float*)d_in[3];
    const float* conv_w = (const float*)d_in[4];
    const float* conv_b = (const float*)d_in[5];
    const float* xprj_w = (const float*)d_in[6];
    const float* dt_w   = (const float*)d_in[7];
    const float* dt_b   = (const float*)d_in[8];
    const float* A_log  = (const float*)d_in[9];
    const float* Dp     = (const float*)d_in[10];
    const float* out_w  = (const float*)d_in[11];
    const float* ln_g   = (const float*)d_in[12];
    const float* ln_b   = (const float*)d_in[13];

    uint8_t* wsp = (uint8_t*)d_ws;
    auto carve = [&](size_t bytes) -> void* {
        void* p = (void*)wsp;
        wsp += (bytes + 255) & ~(size_t)255;
        return p;
    };
    float*    h    = (float*)   carve((size_t)BL * D_MODEL * 4);
    uint16_t* hb   = (uint16_t*)carve((size_t)BL * D_MODEL * 2);
    float*    xz   = (float*)   carve((size_t)BL * 2 * D_INNER * 4);
    float*    xc   = (float*)   carve((size_t)BL * D_INNER * 4);
    uint16_t* xcb  = (uint16_t*)carve((size_t)BL * D_INNER * 2);
    float*    dbc  = (float*)   carve((size_t)BL * 48 * 4);
    float*    dtv  = (float*)   carve((size_t)BL * D_INNER * 4);
    uint16_t* y2b  = (uint16_t*)carve((size_t)BL * D_INNER * 2);
    uint16_t* w_xz = (uint16_t*)carve((size_t)DEPTH * 2 * D_INNER * D_MODEL * 2);
    uint16_t* w_xp = (uint16_t*)carve((size_t)DEPTH * 48 * D_INNER * 2);
    uint16_t* w_dt = (uint16_t*)carve((size_t)DEPTH * D_INNER * DT_RANK * 2);
    uint16_t* w_ot = (uint16_t*)carve((size_t)DEPTH * D_MODEL * D_INNER * 2);

    auto cvt = [&](const float* s, uint16_t* d, int n) {
        k_f32_to_bf16<<<(n + 255) / 256, 256, 0, stream>>>(s, d, n);
    };
    cvt(m_in_w, w_xz, DEPTH * 2 * D_INNER * D_MODEL);
    cvt(xprj_w, w_xp, DEPTH * 48 * D_INNER);
    cvt(dt_w,   w_dt, DEPTH * D_INNER * DT_RANK);
    cvt(out_w,  w_ot, DEPTH * D_MODEL * D_INNER);

    k_in_proj<<<BL, 256, 0, stream>>>(x, in_w, in_b, h, hb);

    for (int layer = 0; layer < DEPTH; ++layer) {
        const uint16_t* wxz = w_xz + (size_t)layer * 2 * D_INNER * D_MODEL;
        const uint16_t* wxp = w_xp + (size_t)layer * 48 * D_INNER;
        const uint16_t* wdt = w_dt + (size_t)layer * D_INNER * DT_RANK;
        const uint16_t* wot = w_ot + (size_t)layer * D_MODEL * D_INNER;

        // xz = h @ m_in_w^T            (65536 x 1024, K=256)
        k_gemm_bf16<4><<<dim3(BL / 128, (2 * D_INNER) / 64), 256, 0, stream>>>(
            hb, wxz, xz, nullptr, D_MODEL, 2 * D_INNER);

        // xc = silu(dwconv(xpart))
        k_conv<<<BL, D_INNER, 0, stream>>>(
            xz, conv_w + (size_t)layer * D_INNER * D_CONV,
            conv_b + (size_t)layer * D_INNER, xc, xcb);

        // dbc = xc @ xproj_w^T         (65536 x 48, K=512)
        k_gemm_bf16<3><<<dim3(BL / 128, 1), 256, 0, stream>>>(
            xcb, wxp, dbc, nullptr, D_INNER, 48);

        // dt = softplus(dbc[:, :16] @ dt_w^T + dt_b)
        k_dt<<<dim3(BL / 128, D_INNER / 64), 256, 0, stream>>>(
            dbc, wdt, dt_b + (size_t)layer * D_INNER, dtv);

        // selective scan + D skip + silu(z) gate -> y2b (bf16)
        k_scan<<<dim3(BATCH, D_INNER / 128), 128, 0, stream>>>(
            xc, dtv, dbc, xz,
            A_log + (size_t)layer * D_INNER * N_STATE,
            Dp + (size_t)layer * D_INNER, y2b);

        // h = y2 @ out_w^T             (65536 x 256, K=512), f32 + bf16 out
        k_gemm_bf16<4><<<dim3(BL / 128, D_MODEL / 64), 256, 0, stream>>>(
            y2b, wot, h, hb, D_INNER, D_MODEL);
    }

    k_ln<<<BL, D_MODEL, 0, stream>>>(h, ln_g, ln_b, (float*)d_out);

    (void)in_sizes; (void)n_in; (void)out_size; (void)ws_size;
}
